// Cace_7155415515517
// MI455X (gfx1250) — compile-verified, matrix operand order checked by reading, not run on hardware
//
#include <hip/hip_runtime.h>

typedef __attribute__((ext_vector_type(16))) _Float16 v16h;
typedef __attribute__((ext_vector_type(8)))  float    v8f;

#define NB      3
#define NCHAN   9
#define NRBF    8
#define NANG    20
#define RAW     160     // NRBF * NANG
#define BOUT    288     // NRBF * 4 * NCHAN
#define CUTOFF_F 5.5f

// (lx,ly,lz) tables in reference order, and multinomial prefactors l!/(lx!ly!lz!)
__constant__ int   c_lx[NANG]  = {0, 1,0,0, 2,1,1,0,0,0, 3,2,2,1,1,1,0,0,0,0};
__constant__ int   c_ly[NANG]  = {0, 0,1,0, 0,1,0,2,1,0, 0,1,0,2,1,0,3,2,1,0};
__constant__ int   c_lz[NANG]  = {0, 0,0,1, 0,0,1,0,1,2, 0,0,1,0,1,2,0,1,2,3};
__constant__ float c_pref[NANG]= {1.f, 1.f,1.f,1.f, 1.f,2.f,2.f,1.f,2.f,1.f,
                                  1.f,3.f,3.f,3.f,6.f,3.f,1.f,3.f,3.f,1.f};

// ---------------------------------------------------------------- zero scratch
__global__ void zero_ws(float4* __restrict__ p, long n4) {
    long i = (long)blockIdx.x * blockDim.x + threadIdx.x;
    if (i < n4) p[i] = make_float4(0.f, 0.f, 0.f, 0.f);
}

// ------------------------------------------------- edge kernel: one wave/edge
// RA[s][node][r*20+a] += radial[r]*ang[a]   (s = species of src node)
__global__ void edge_scatter(const int*   __restrict__ eidx,
                             const float* __restrict__ elen,
                             const float* __restrict__ evec,
                             const int*   __restrict__ zn,
                             float*       __restrict__ RA,
                             int n_edges, int n_nodes) {
    int gid  = blockIdx.x * blockDim.x + threadIdx.x;
    int e    = gid >> 5;
    int lane = threadIdx.x & 31;
    if (e >= n_edges) return;

    int src = eidx[e];
    int dst = eidx[n_edges + e];
    int s   = zn[src] & 1;

    float r  = elen[e];
    float vx = evec[3*e+0], vy = evec[3*e+1], vz = evec[3*e+2];
    float inv = rsqrtf(vx*vx + vy*vy + vz*vz);
    float ux = vx*inv, uy = vy*inv, uz = vz*inv;

    // polynomial cutoff (p=6): 1 - 28u^6 + 48u^7 - 21u^8
    float u  = r * (1.0f / CUTOFF_F);
    float u2 = u*u, u3 = u2*u, u6 = u3*u3, u7 = u6*u, u8 = u7*u;
    float fc = (u < 1.f) ? (1.f - 28.f*u6 + 48.f*u7 - 21.f*u8) : 0.f;
    float amp = sqrtf(2.f / CUTOFF_F) * fc / r;

    // sin(k*pi*r/C), k=1..8 via Chebyshev recurrence
    float t  = 3.14159265358979f * u;
    float s1 = __sinf(t), c1 = __cosf(t);
    float radial[NRBF];
    radial[0] = s1;
    float sm1 = 0.f, sk = s1, twoC = 2.f*c1;
    #pragma unroll
    for (int k = 1; k < NRBF; ++k) { float sn = twoC*sk - sm1; sm1 = sk; sk = sn; radial[k] = sn; }
    #pragma unroll
    for (int k = 0; k < NRBF; ++k) radial[k] *= amp;

    // angular term for this lane (lane == angular index a)
    float ux2=ux*ux, ux3=ux2*ux, uy2=uy*uy, uy3=uy2*uy, uz2=uz*uz, uz3=uz2*uz;
    if (lane < NANG) {
        int lx = c_lx[lane], ly = c_ly[lane], lz = c_lz[lane];
        float px = (lx==0)?1.f:((lx==1)?ux:((lx==2)?ux2:ux3));
        float py = (ly==0)?1.f:((ly==1)?uy:((ly==2)?uy2:uy3));
        float pz = (lz==0)?1.f:((lz==1)?uz:((lz==2)?uz2:uz3));
        float ang = px*py*pz;
        float* base = RA + ((size_t)s * n_nodes + dst) * RAW + lane;
        #pragma unroll
        for (int rr = 0; rr < NRBF; ++rr)
            unsafeAtomicAdd(base + rr*NANG, radial[rr]*ang);   // -> global_atomic_add_f32
    }
}

// ------------------------------------------- node kernel: one wave/node, WMMA
// B_node[32x9] = Gm[32x5] x Em[5x9] done as two v_wmma_f32_16x16x32_f16
__global__ void node_wmma(const int*   __restrict__ zn,
                          const float* __restrict__ W,
                          const float* __restrict__ RA,
                          float*       __restrict__ out,
                          int n_nodes) {
    int gid  = blockIdx.x * blockDim.x + threadIdx.x;
    int n    = gid >> 5;
    int lane = threadIdx.x & 31;
    if (n >= n_nodes) return;

    // lane owns (r,l) pair p = lane
    int r = lane >> 2;
    int l = lane & 3;
    const float* ra0 = RA + (size_t)n             * RAW + r*NANG;
    const float* ra1 = RA + ((size_t)n_nodes + n) * RAW + r*NANG;

    float g0=0.f, g1=0.f, g2=0.f, g3=0.f, g4=0.f;
    if (l == 0) {
        g0 = ra0[0]; g1 = ra1[0];
    } else {
        int start = (l==1) ? 1 : ((l==2) ? 4 : 10);
        int cnt   = (l==1) ? 3 : ((l==2) ? 6 : 10);
        float G00=0.f, G01=0.f, G11=0.f;
        for (int k = 0; k < cnt; ++k) {
            int a = start + k;
            float pf = c_pref[a];
            float x0 = ra0[a], x1 = ra1[a];
            G00 += pf*x0*x0; G01 += pf*x0*x1; G11 += pf*x1*x1;
        }
        g2 = G00; g3 = 2.f*G01; g4 = G11;
    }

    // B operand: column c = channel (lanes 0-15), rows K=0..4 = {e0,e1,e0^2,e0*e1,e1^2}
    int z = zn[n] & 1;
    int c = lane & 15;
    float e0 = 0.f, e1 = 0.f;
    if (lane < 16 && c < NCHAN) {
        int i = c / 3, j = c - 3*i;
        float wj = W[z*NB + j];
        e0 = W[0*NB + i] * wj;
        e1 = W[1*NB + i] * wj;
    }
    v16h b = {};
    b[0] = (_Float16)e0;
    b[1] = (_Float16)e1;
    b[2] = (_Float16)(e0*e0);
    b[3] = (_Float16)(e0*e1);     // pairs with g3 = 2*G01
    b[4] = (_Float16)(e1*e1);

    bool lo = lane < 16;
    // A tile 0: M rows = pairs 0..15 (already in lanes 0..15)
    v16h a0 = {};
    a0[0] = (_Float16)(lo ? g0 : 0.f);
    a0[1] = (_Float16)(lo ? g1 : 0.f);
    a0[2] = (_Float16)(lo ? g2 : 0.f);
    a0[3] = (_Float16)(lo ? g3 : 0.f);
    a0[4] = (_Float16)(lo ? g4 : 0.f);
    // A tile 1: M rows = pairs 16..31 -> shuffle down from lanes 16..31
    float h0 = __shfl(g0, lane + 16, 32);
    float h1 = __shfl(g1, lane + 16, 32);
    float h2 = __shfl(g2, lane + 16, 32);
    float h3 = __shfl(g3, lane + 16, 32);
    float h4 = __shfl(g4, lane + 16, 32);
    v16h a1 = {};
    a1[0] = (_Float16)(lo ? h0 : 0.f);
    a1[1] = (_Float16)(lo ? h1 : 0.f);
    a1[2] = (_Float16)(lo ? h2 : 0.f);
    a1[3] = (_Float16)(lo ? h3 : 0.f);
    a1[4] = (_Float16)(lo ? h4 : 0.f);

    v8f cz = {};
    v8f d0 = __builtin_amdgcn_wmma_f32_16x16x32_f16(false, a0, false, b, (short)0, cz, false, false);
    v8f d1 = __builtin_amdgcn_wmma_f32_16x16x32_f16(false, a1, false, b, (short)0, cz, false, false);

    // D layout: lane L -> column N = L&15 ; VGPR v -> row M = v + 8*(L>=16)
    if (c < NCHAN) {
        float* o  = out + (size_t)n * BOUT;
        int   mhi = (lane >> 4) & 1;
        #pragma unroll
        for (int v = 0; v < 8; ++v) {
            int m = v + mhi*8;
            o[m*NCHAN + c]        = d0[v];   // pairs 0..15
            o[(16 + m)*NCHAN + c] = d1[v];   // pairs 16..31
        }
    }
}

// ------------------------------------------------------------------- launcher
extern "C" void kernel_launch(void* const* d_in, const int* in_sizes, int n_in,
                              void* d_out, int out_size, void* d_ws, size_t ws_size,
                              hipStream_t stream) {
    const int*   zn   = (const int*)  d_in[1];   // atomic_numbers
    const int*   eidx = (const int*)  d_in[2];   // edge_index [2, E]
    const float* elen = (const float*)d_in[3];   // edge_lengths
    const float* evec = (const float*)d_in[4];   // edge_vectors [E,3]
    const float* W    = (const float*)d_in[5];   // W_embed [2,3]

    int n_nodes = in_sizes[1];
    int n_edges = in_sizes[3];

    float* RA = (float*)d_ws;                    // [2, n_nodes, 160]
    long n4 = ((long)2 * n_nodes * RAW) / 4;
    zero_ws<<<(unsigned)((n4 + 255) / 256), 256, 0, stream>>>((float4*)d_ws, n4);

    long eth = (long)n_edges * 32;
    edge_scatter<<<(unsigned)((eth + 255) / 256), 256, 0, stream>>>(
        eidx, elen, evec, zn, RA, n_edges, n_nodes);

    long nth = (long)n_nodes * 32;
    node_wmma<<<(unsigned)((nth + 255) / 256), 256, 0, stream>>>(
        zn, W, RA, (float*)d_out, n_nodes);
}
